// PointNet2Encoder_20718922236583
// MI455X (gfx1250) — matrix operand order repaired and checked by reading, not compiled, not run on hardware
//
#include <hip/hip_runtime.h>
#include <hip/hip_bf16.h>

// ---------------------------------------------------------------------------
// PointNet++ MSG encoder for gfx1250 (MI455X).
//  - FPS: 1 block/batch, mind[] in registers, shuffle+LDS argmax (lowest-index ties).
//  - Ball query: 1 wave/center, ballot/popc sequential scan (matches argsort-by-index).
//  - MLPs: f16 WMMA 16x16x32, f32 accum. Weights staged in LDS pre-swizzled into
//    B-fragment order (one contiguous 32B read per lane); A fragments are two
//    contiguous 16B LDS reads per the ISA 7.12.2 layout. Activations ping-pong
//    in LDS per wave; layer-3 max-pool in registers via shfl_xor.
//  - Final 4096x256 @ 256x125 linear: 1 wave per 16x16 tile, WMMA, N padded to
//    128 via clamp+mask (no exec divergence around loads).
// ---------------------------------------------------------------------------

typedef __attribute__((ext_vector_type(16))) _Float16 v16h;
typedef __attribute__((ext_vector_type(8)))  _Float16 v8h;
typedef __attribute__((ext_vector_type(8)))  float    v8f;

#define DEVINL __device__ __forceinline__

DEVINL v8f wmma_f16(v16h a, v16h b, v8f c) {
  return __builtin_amdgcn_wmma_f32_16x16x32_f16(
      /*neg_a=*/false, a, /*neg_b=*/false, b,
      /*c_mod=*/(short)0, c, /*reuse_a=*/false, /*reuse_b=*/false);
}

// A fragment: 16x32 f16 tile from LDS (row-major, leading dim `ld`).
// Per ISA 7.12.2 the 16 halves of lane (hs,nloc) are two contiguous 8-half
// runs: K = kt*32 + hs*8 + [0..7] and + 16 + [0..7]  ->  2x ds_load_b128.
DEVINL v16h load_frag_a(const _Float16* act, int ld, int mt, int kt, int lane) {
  const int m  = mt * 16 + (lane & 15);
  const int hs = (lane >> 4) & 1;
  const _Float16* row = act + m * ld + kt * 32 + hs * 8;
  v8h lo = *(const v8h*)(row);
  v8h hi = *(const v8h*)(row + 16);
  return __builtin_shufflevector(lo, hi, 0, 1, 2, 3, 4, 5, 6, 7,
                                 8, 9, 10, 11, 12, 13, 14, 15);
}

// B fragment from pre-swizzled LDS weights: layout [kt][nt][lane][16 halves],
// so a lane's fragment is one contiguous 32-byte chunk (2x ds_load_b128).
DEVINL v16h load_frag_b_sw(const _Float16* w, int NT, int kt, int nt, int lane) {
  const _Float16* p = w + ((((kt * NT + nt) * 32) + lane) << 4);
  return *(const v16h*)p;
}

// ---------------------------------------------------------------------------
// Farthest point sampling. One block per batch; NP sequential steps.
// Emits centroid coords directly (idx[0]=0, then successive argmaxes).
// ---------------------------------------------------------------------------
template<int N, int NP, int XS>
__global__ __launch_bounds__(1024)
void fps_kernel(const float* __restrict__ xyz, float* __restrict__ new_xyz) {
  constexpr int PT = N / 1024;
  __shared__ float redv[32];
  __shared__ int   redi[32];
  __shared__ int   s_far;
  const int b = blockIdx.x, t = threadIdx.x;
  const float* xb = xyz + (size_t)b * N * XS;
  float mind[PT];
#pragma unroll
  for (int i = 0; i < PT; ++i) mind[i] = 1e10f;
  int far = 0;
  for (int s = 0; s < NP; ++s) {
    const float cx = xb[far * XS + 0], cy = xb[far * XS + 1], cz = xb[far * XS + 2];
    if (t == 0) {
      float* o = new_xyz + ((size_t)b * NP + s) * 3;
      o[0] = cx; o[1] = cy; o[2] = cz;
    }
    float bv = -1.0f; int bi = t * PT;
#pragma unroll
    for (int i = 0; i < PT; ++i) {
      const int p = t * PT + i;
      float dx = xb[p * XS + 0] - cx, dy = xb[p * XS + 1] - cy, dz = xb[p * XS + 2] - cz;
      float d  = dx * dx + dy * dy + dz * dz;
      float mm = fminf(mind[i], d);
      mind[i] = mm;
      if (mm > bv) { bv = mm; bi = p; }
    }
#pragma unroll
    for (int off = 16; off > 0; off >>= 1) {
      float ov = __shfl_down(bv, off);
      int   oi = __shfl_down(bi, off);
      if (ov > bv || (ov == bv && oi < bi)) { bv = ov; bi = oi; }
    }
    if ((t & 31) == 0) { redv[t >> 5] = bv; redi[t >> 5] = bi; }
    __syncthreads();
    if (t < 32) {
      bv = redv[t]; bi = redi[t];
#pragma unroll
      for (int off = 16; off > 0; off >>= 1) {
        float ov = __shfl_down(bv, off);
        int   oi = __shfl_down(bi, off);
        if (ov > bv || (ov == bv && oi < bi)) { bv = ov; bi = oi; }
      }
      if (t == 0) s_far = bi;
    }
    __syncthreads();
    far = s_far;
  }
}

// ---------------------------------------------------------------------------
// One MLP layer on a per-wave activation tile: out = relu(in @ W * s + b).
// All channel counts are multiples of 16; KPAD multiple of 32 (zero padded).
// ---------------------------------------------------------------------------
template<int ROWS, int KPAD, int COUT, int LD>
DEVINL void mlp_layer(const _Float16* aIn, _Float16* aOut, const _Float16* wl,
                      const float* __restrict__ sg, const float* __restrict__ bg, int lane) {
  static_assert(ROWS % 16 == 0 && COUT % 16 == 0 && KPAD % 32 == 0 && LD % 8 == 0, "tile shapes");
  {
    v8h zz = {};
    v8h* dst = (v8h*)aOut;
    for (int i = lane; i < ROWS * LD / 8; i += 32) dst[i] = zz;
  }
  constexpr int MT = ROWS / 16, NT = COUT / 16, KT = KPAD / 32;
  const int nloc = lane & 15, hs = (lane >> 4) & 1;
#pragma unroll
  for (int mt = 0; mt < MT; ++mt)
#pragma unroll
    for (int nt = 0; nt < NT; ++nt) {
      v8f acc = {};
#pragma unroll
      for (int kt = 0; kt < KT; ++kt) {
        v16h a = load_frag_a(aIn, LD, mt, kt, lane);
        v16h w = load_frag_b_sw(wl, NT, kt, nt, lane);
        acc = wmma_f16(a, w, acc);
      }
      const int n = nt * 16 + nloc;
      const float sc = sg[n], bi = bg[n];
#pragma unroll
      for (int r = 0; r < 8; ++r) {
        float v = acc[r] * sc + bi;
        v = v > 0.0f ? v : 0.0f;
        aOut[(mt * 16 + r + hs * 8) * LD + n] = (_Float16)v;
      }
    }
}

// Final MLP layer fused with max-pool over the ROWS (sample) axis.
template<int ROWS, int KPAD, int COUT, int LD>
DEVINL void mlp_layer_max(const _Float16* aIn, const _Float16* wl,
                          const float* __restrict__ sg, const float* __restrict__ bg,
                          float* __restrict__ outp, int lane) {
  static_assert(ROWS % 16 == 0 && COUT % 16 == 0 && KPAD % 32 == 0, "tile shapes");
  constexpr int MT = ROWS / 16, NT = COUT / 16, KT = KPAD / 32;
  const int nloc = lane & 15, hs = (lane >> 4) & 1;
#pragma unroll
  for (int nt = 0; nt < NT; ++nt) {
    const int n = nt * 16 + nloc;
    const float sc = sg[n], bi = bg[n];
    float mx = -3.4e38f;
#pragma unroll
    for (int mt = 0; mt < MT; ++mt) {
      v8f acc = {};
#pragma unroll
      for (int kt = 0; kt < KT; ++kt) {
        v16h a = load_frag_a(aIn, LD, mt, kt, lane);
        v16h w = load_frag_b_sw(wl, NT, kt, nt, lane);
        acc = wmma_f16(a, w, acc);
      }
#pragma unroll
      for (int r = 0; r < 8; ++r) {
        float v = acc[r] * sc + bi;
        v = v > 0.0f ? v : 0.0f;
        mx = fmaxf(mx, v);
      }
    }
    mx = fmaxf(mx, __shfl_xor(mx, 16));
    if (hs == 0) outp[n] = mx;
  }
}

// ---------------------------------------------------------------------------
// Fused ball-query + gather + 3-layer MLP + max-pool. One wave per center,
// 4 waves per block; weights staged (pre-swizzled) in LDS once per block.
// ---------------------------------------------------------------------------
template<int NPTS, int NCENT, int K_, int CIN, int CINPAD, int C1, int C2, int C3,
         int LD, int XS, int FS, int OUTC>
__global__ __launch_bounds__(128)
void sa_branch_kernel(const float* __restrict__ xyz,   // B x NPTS x XS (xyz = first 3)
                      const float* __restrict__ feats, // B x NPTS x FS (base pre-offset)
                      const float* __restrict__ cxyz,  // B x NCENT x 3 centers
                      const float* __restrict__ W1, const float* __restrict__ s1, const float* __restrict__ b1,
                      const float* __restrict__ W2, const float* __restrict__ s2, const float* __restrict__ b2,
                      const float* __restrict__ W3, const float* __restrict__ s3, const float* __restrict__ b3,
                      float* __restrict__ outf, int choff, float radius) {
  constexpr int C1PAD = (C1 + 31) / 32 * 32;
  constexpr int C2PAD = (C2 + 31) / 32 * 32;
  constexpr int NW = 4;
  extern __shared__ char smem_raw[];
  _Float16* wl1  = (_Float16*)smem_raw;
  _Float16* wl2  = wl1 + CINPAD * C1;
  _Float16* wl3  = wl2 + C1PAD * C2;
  _Float16* acts = wl3 + C2PAD * C3;
  int*      idxs = (int*)(acts + NW * 2 * K_ * LD);

  const int tid = threadIdx.x, lane = tid & 31, warp = tid >> 5;

  // Stage weights f32 -> f16, zero-padding K to a multiple of 32, and swizzle
  // into B-fragment order: dst[((kt*NT + nt)*32 + (hs*16+nloc)) * 16 + e]
  // where kk = kt*32 + (e>=8 ? 16 : 0) + hs*8 + (e&7), n = nt*16 + nloc.
  auto stage = [&](const float* g, _Float16* l, int Ks, int Kp, int Nc) {
    const int NTl = Nc >> 4;
    for (int i = tid; i < Kp * Nc; i += 128) {
      const int kk = i / Nc, n = i - kk * Nc;
      const int kt = kk >> 5, kl = kk & 31;
      const int hsw = (kl >> 3) & 1, e = (kl & 7) + ((kl >> 4) << 3);
      const int nt = n >> 4, nloc = n & 15;
      const float v = (kk < Ks) ? g[(size_t)kk * Nc + n] : 0.0f;
      l[((((kt * NTl + nt) << 5) + (hsw * 16 + nloc)) << 4) + e] = (_Float16)v;
    }
  };
  stage(W1, wl1, CIN, CINPAD, C1);
  stage(W2, wl2, C1, C1PAD, C2);
  stage(W3, wl3, C2, C2PAD, C3);
  __syncthreads();

  _Float16* act0 = acts + warp * (2 * K_ * LD);
  _Float16* act1 = act0 + K_ * LD;
  int* idxbuf = idxs + warp * K_;

  const int cg = blockIdx.x * NW + warp;
  const int b  = cg / NCENT;
  const int ci = cg % NCENT;
  const float* cc = cxyz + ((size_t)b * NCENT + ci) * 3;
  const float cx = cc[0], cy = cc[1], cz = cc[2];
  const float r2 = radius * radius;

  // Ball query: first K_ indices (ascending) with d2 < r2; pad with first hit.
  int found = 0;
  for (int base = 0; base < NPTS && found < K_; base += 32) {
    const int p = base + lane;
    const float* q = xyz + ((size_t)b * NPTS + p) * XS;
    float dx = q[0] - cx, dy = q[1] - cy, dz = q[2] - cz;
    bool in = (dx * dx + dy * dy + dz * dz) < r2;
    unsigned mk = (unsigned)__ballot(in);
    int pos = found + __popc(mk & ((1u << lane) - 1u));
    if (in && pos < K_) idxbuf[pos] = p;
    found += __popc(mk);
  }
  if (found == 0) { if (lane == 0) idxbuf[0] = 0; found = 1; }
  if (found > K_) found = K_;
  const int first = idxbuf[0];
  for (int j = lane; j < K_; j += 32)
    if (j >= found) idxbuf[j] = first;

  // Gather: rows = samples, cols = [xyz - center (3), feats (CIN-3)], zero pad.
  for (int i = lane; i < K_ * CINPAD; i += 32) {
    const int j = i / CINPAD, c = i - j * CINPAD;
    const int p = idxbuf[j];
    float v = 0.0f;
    if (c < 3)        v = xyz[((size_t)b * NPTS + p) * XS + c] - (c == 0 ? cx : (c == 1 ? cy : cz));
    else if (c < CIN) v = feats[((size_t)b * NPTS + p) * FS + (c - 3)];
    act0[j * LD + c] = (_Float16)v;
  }

  mlp_layer<K_, CINPAD, C1, LD>(act0, act1, wl1, s1, b1, lane);
  mlp_layer<K_, C1PAD,  C2, LD>(act1, act0, wl2, s2, b2, lane);
  mlp_layer_max<K_, C2PAD, C3, LD>(act0, wl3, s3, b3,
      outf + ((size_t)b * NCENT + ci) * OUTC + choff, lane);
}

// ---------------------------------------------------------------------------
// Final linear: out[:, 3:128] = feats2(4096x256) @ linW(256x125) + linb,
// out[:, 0:3] = new_xyz2. One wave per 16x16 tile; N padded to 128 with a
// clamped index + 0/1 mask so every load is unconditional (no exec churn).
// ---------------------------------------------------------------------------
__global__ __launch_bounds__(128)
void final_linear_kernel(const float* __restrict__ f2, const float* __restrict__ linW,
                         const float* __restrict__ linb, const float* __restrict__ nx2,
                         float* __restrict__ out) {
  const int lane = threadIdx.x & 31;
  const int wid  = blockIdx.x * 4 + (threadIdx.x >> 5);
  const int mt = wid >> 3, nt = wid & 7;
  const int nloc = lane & 15, hs = (lane >> 4) & 1;
  const int m = mt * 16 + nloc;  // A-fragment row for this lane
  const int n = nt * 16 + nloc;
  const int   nc  = (n < 125) ? n : 124;   // valid address for padded columns
  const float msk = (n < 125) ? 1.0f : 0.0f;
  v8f acc = {};
#pragma unroll
  for (int kt = 0; kt < 8; ++kt) {
    v16h a, w;
#pragma unroll
    for (int e = 0; e < 16; ++e) {
      int kk = kt * 32 + hs * 8 + ((e & 8) ? 16 : 0) + (((e >> 1) & 3) << 1) + (e & 1);
      a[e] = (_Float16)f2[(size_t)m * 256 + kk];
    }
#pragma unroll
    for (int e = 0; e < 16; ++e) {
      int kk = kt * 32 + hs * 8 + ((e & 8) ? 16 : 0) + (e & 7);
      w[e] = (_Float16)(linW[(size_t)kk * 125 + nc] * msk);
    }
    acc = wmma_f16(a, w, acc);
  }
  if (n < 125) {
    const float bi = linb[n];
#pragma unroll
    for (int r = 0; r < 8; ++r) {
      const int row = mt * 16 + r + hs * 8;
      out[(size_t)row * 128 + 3 + n] = acc[r] + bi;
    }
  }
  if (nt == 0 && nloc < 3) {
#pragma unroll
    for (int r = 0; r < 8; ++r) {
      const int row = mt * 16 + r + hs * 8;
      out[(size_t)row * 128 + nloc] = nx2[(size_t)row * 3 + nloc];
    }
  }
}

// ---------------------------------------------------------------------------
static constexpr size_t sa_lds_bytes(int CINPAD, int C1, int C2, int C3, int K_, int LD) {
  const int C1P = (C1 + 31) / 32 * 32, C2P = (C2 + 31) / 32 * 32;
  const size_t halves = (size_t)CINPAD * C1 + (size_t)C1P * C2 + (size_t)C2P * C3
                      + (size_t)4 * 2 * K_ * LD;
  return halves * 2 + (size_t)4 * K_ * 4;
}

extern "C" void kernel_launch(void* const* d_in, const int* in_sizes, int n_in,
                              void* d_out, int out_size, void* d_ws, size_t ws_size,
                              hipStream_t stream) {
  (void)in_sizes; (void)n_in; (void)out_size; (void)ws_size;
  // Input order (setup_inputs insertion order):
  //  0: pointcloud (16,8192,9)
  //  1..9 : sa1 mlp0 (W,s,b) x3 ; 10..18: sa1 mlp1 ; 19..27: sa2 mlp0 ; 28..36: sa2 mlp1
  //  37: linW (256,125) ; 38: linb (125)
  const float* pc = (const float*)d_in[0];
  auto P = [&](int i) { return (const float*)d_in[i]; };

  float* nx1 = (float*)d_ws;                       // 16*1024*3
  float* f1  = nx1 + (size_t)16 * 1024 * 3;        // 16*1024*96
  float* nx2 = f1  + (size_t)16 * 1024 * 96;       // 16*256*3
  float* f2  = nx2 + (size_t)16 * 256 * 3;         // 16*256*256
  float* out = (float*)d_out;

  // ---- SA1 ----
  fps_kernel<8192, 1024, 9><<<16, 1024, 0, stream>>>(pc, nx1);

  sa_branch_kernel<8192, 1024, 16, 9, 32, 16, 16, 32, 32, 9, 9, 96>
      <<<16 * 1024 / 4, 128, sa_lds_bytes(32, 16, 16, 32, 16, 32), stream>>>(
          pc, pc + 3, nx1,
          P(1), P(2), P(3), P(4), P(5), P(6), P(7), P(8), P(9),
          f1, 0, 0.05f);

  sa_branch_kernel<8192, 1024, 32, 9, 32, 32, 32, 64, 64, 9, 9, 96>
      <<<16 * 1024 / 4, 128, sa_lds_bytes(32, 32, 32, 64, 32, 64), stream>>>(
          pc, pc + 3, nx1,
          P(10), P(11), P(12), P(13), P(14), P(15), P(16), P(17), P(18),
          f1, 32, 0.1f);

  // ---- SA2 ----
  fps_kernel<1024, 256, 3><<<16, 1024, 0, stream>>>(nx1, nx2);

  sa_branch_kernel<1024, 256, 16, 99, 128, 64, 64, 128, 128, 3, 96, 256>
      <<<16 * 256 / 4, 128, sa_lds_bytes(128, 64, 64, 128, 16, 128), stream>>>(
          nx1, f1, nx2,
          P(19), P(20), P(21), P(22), P(23), P(24), P(25), P(26), P(27),
          f2, 0, 0.1f);

  sa_branch_kernel<1024, 256, 32, 99, 128, 64, 96, 128, 128, 3, 96, 256>
      <<<16 * 256 / 4, 128, sa_lds_bytes(128, 64, 96, 128, 32, 128), stream>>>(
          nx1, f1, nx2,
          P(28), P(29), P(30), P(31), P(32), P(33), P(34), P(35), P(36),
          f2, 128, 0.2f);

  // ---- head: concat(xyz2, feats2 @ linW + linb) ----
  final_linear_kernel<<<(256 * 8) / 4, 128, 0, stream>>>(f2, P(37), P(38), nx2, out);
}
